// GNNNodeEmbed_163208757329
// MI455X (gfx1250) — matrix operand order repaired and checked by reading, not compiled, not run on hardware
//
#include <hip/hip_runtime.h>
#include <hip/hip_bf16.h>
#include <math.h>

#define HDIM 256

typedef __attribute__((ext_vector_type(16))) _Float16 v16h;
typedef __attribute__((ext_vector_type(8)))  float    v8f;

// ---------------------------------------------------------------------------
// Weight prep: f32 row-major W[k][n] -> f16 transposed Wt[n][k] for all 9
// matrices (enc, W1[2][2], W2[2][2]). Makes WMMA B-fragment loads contiguous.
// ---------------------------------------------------------------------------
__global__ __launch_bounds__(256) void prep_weights_kernel(
    const float* __restrict__ encW, const float* __restrict__ W1,
    const float* __restrict__ W2, _Float16* __restrict__ wt, int total)
{
    int gid = blockIdx.x * blockDim.x + threadIdx.x;
    if (gid >= total) return;
    int mat = gid >> 16;            // H*H = 65536 elements per matrix
    int idx = gid & 65535;
    int n = idx >> 8;
    int k = idx & 255;
    const float* src;
    if (mat == 0)       src = encW;
    else if (mat <= 4)  src = W1 + (size_t)(mat - 1) * HDIM * HDIM;
    else                src = W2 + (size_t)(mat - 5) * HDIM * HDIM;
    wt[(size_t)mat * HDIM * HDIM + (size_t)n * HDIM + k] = (_Float16)src[(size_t)k * HDIM + n];
}

// ---------------------------------------------------------------------------
// GEMM: out[nrows x 256] = act( A[nrows x 256] @ W[256 x 256] + bias )
// Wt is f16, transposed (Wt[n*256 + k]). A is f32, converted to f16 in regs.
// Each wave computes a 16x64 strip: 4 f32 accumulators, 8 K-steps of
// v_wmma_f32_16x16x32_f16 -> 32 WMMAs per wave. act: 0=none, 1=relu, 2=elu.
// ---------------------------------------------------------------------------
__global__ __launch_bounds__(256) void gemm_wmma_kernel(
    const float* __restrict__ A, const _Float16* __restrict__ Wt,
    const float* __restrict__ bias, float* __restrict__ out,
    int nrows, int act)
{
    const int lane      = threadIdx.x & 31;
    const int waveInBlk = threadIdx.x >> 5;
    const int waveId    = blockIdx.x * (blockDim.x >> 5) + waveInBlk;
    const int tiles_m   = (nrows + 15) >> 4;
    const int m_tile    = waveId >> 2;        // 4 n-blocks of 64 cover H=256
    const int nblock    = (waveId & 3) * 64;
    if (m_tile >= tiles_m) return;            // wave-uniform: EXEC stays all-1s

    const int half_sel = lane >> 4;           // 0: lanes 0-15, 1: lanes 16-31
    const int mrow     = m_tile * 16 + (lane & 15);
    const int mclamp   = mrow < nrows ? mrow : nrows - 1;
    const int asel     = half_sel * 16;       // A-fragment K sub-offset (contiguous k = 16h + s)
    const int bsel     = half_sel * 16;       // B-fragment K sub-offset (contiguous k = 16h + s)
    const int ncol     = nblock + (lane & 15);

    v8f acc[4];
#pragma unroll
    for (int j = 0; j < 4; ++j) {
        float bv = bias[nblock + 16 * j + (lane & 15)];
#pragma unroll
        for (int r = 0; r < 8; ++r) acc[j][r] = bv;   // bias depends only on col
    }

    const float* arow = A + (size_t)mclamp * HDIM;

#pragma unroll
    for (int k0 = 0; k0 < HDIM; k0 += 32) {
        // A fragment: elems 0..7 = K (k0+asel ..), elems 8..15 = K (k0+asel+8 ..)
        v8f alo = *(const v8f*)(arow + k0 + asel);
        v8f ahi = *(const v8f*)(arow + k0 + asel + 8);
        v16h afrag;
#pragma unroll
        for (int i = 0; i < 8; ++i) {
            afrag[i]     = (_Float16)alo[i];
            afrag[i + 8] = (_Float16)ahi[i];
        }
#pragma unroll
        for (int j = 0; j < 4; ++j) {
            // B fragment: lane = column, 16 consecutive K values (contiguous 32B)
            v16h bfrag = *(const v16h*)(Wt + (size_t)(ncol + 16 * j) * HDIM + k0 + bsel);
            acc[j] = __builtin_amdgcn_wmma_f32_16x16x32_f16(
                false, afrag, false, bfrag, (short)0, acc[j], false, false);
        }
    }

    // C/D layout: VGPR r -> row m_tile*16 + r + 8*half_sel, col = lane&15 block
#pragma unroll
    for (int j = 0; j < 4; ++j) {
        int col = nblock + 16 * j + (lane & 15);
#pragma unroll
        for (int r = 0; r < 8; ++r) {
            int row = m_tile * 16 + r + half_sel * 8;
            if (row < nrows) {
                float v = acc[j][r];
                if (act == 1)      v = v > 0.0f ? v : 0.0f;
                else if (act == 2) v = v > 0.0f ? v : (expf(v) - 1.0f);
                out[(size_t)row * HDIM + col] = v;
            }
        }
    }
}

// ---------------------------------------------------------------------------
// agg = h  (float4 copy; folds the "+ h" residual before scatter-accumulate)
// ---------------------------------------------------------------------------
__global__ __launch_bounds__(256) void copy4_kernel(
    const float4* __restrict__ in, float4* __restrict__ out, int n4)
{
    int gid = blockIdx.x * blockDim.x + threadIdx.x;
    if (gid < n4) out[gid] = in[gid];
}

// ---------------------------------------------------------------------------
// Edge scatter: agg[dst] += h[src] * attn   (one thread per edge x float4)
// Feature table (51 MB) is L2-resident -> gather + f32 atomics hit L2.
// ---------------------------------------------------------------------------
__global__ __launch_bounds__(256) void scatter_kernel(
    const float4* __restrict__ h, const int* __restrict__ src,
    const int* __restrict__ dst, const float* __restrict__ attn,
    float* __restrict__ agg, int E)
{
    int gid = blockIdx.x * blockDim.x + threadIdx.x;
    int e = gid >> 6;            // H/4 = 64 float4 chunks per row
    int c = gid & 63;
    if (e >= E) return;
    float a = attn[e];
    float4 v = h[(size_t)src[e] * 64 + c];
    float* p = agg + (size_t)dst[e] * HDIM + c * 4;
    __hip_atomic_fetch_add(p + 0, v.x * a, __ATOMIC_RELAXED, __HIP_MEMORY_SCOPE_AGENT);
    __hip_atomic_fetch_add(p + 1, v.y * a, __ATOMIC_RELAXED, __HIP_MEMORY_SCOPE_AGENT);
    __hip_atomic_fetch_add(p + 2, v.z * a, __ATOMIC_RELAXED, __HIP_MEMORY_SCOPE_AGENT);
    __hip_atomic_fetch_add(p + 3, v.w * a, __ATOMIC_RELAXED, __HIP_MEMORY_SCOPE_AGENT);
}

// ---------------------------------------------------------------------------
extern "C" void kernel_launch(void* const* d_in, const int* in_sizes, int n_in,
                              void* d_out, int out_size, void* d_ws, size_t ws_size,
                              hipStream_t stream)
{
    (void)n_in; (void)out_size; (void)ws_size;
    const float* x_a     = (const float*)d_in[0];
    const float* x_b     = (const float*)d_in[1];
    const int*   ei_aa   = (const int*)d_in[2];
    const int*   ei_bb   = (const int*)d_in[3];
    const float* attn_aa = (const float*)d_in[4];
    const float* attn_bb = (const float*)d_in[5];
    const float* enc_W   = (const float*)d_in[6];
    const float* enc_b   = (const float*)d_in[7];
    const float* W1      = (const float*)d_in[8];
    const float* b1      = (const float*)d_in[9];
    const float* W2      = (const float*)d_in[10];
    const float* b2      = (const float*)d_in[11];

    const int H = HDIM;
    const int N = in_sizes[0] / H;
    const int E = in_sizes[4];
    const int T = 2;
    const int L = in_sizes[8] / (T * H * H);
    const size_t NH = (size_t)N * H;

    // Workspace layout (f32 unless noted):
    //   h[2*NH] | agg[NH] | mid[NH] | wt16[9*H*H] (f16)
    float*    h   = (float*)d_ws;
    float*    agg = h + 2 * NH;
    float*    mid = agg + NH;
    _Float16* wt  = (_Float16*)(mid + NH);

    // 1) one-time weight transpose + f32->f16
    int wtot = 9 * H * H;
    prep_weights_kernel<<<(wtot + 255) / 256, 256, 0, stream>>>(enc_W, W1, W2, wt, wtot);

    auto launch_gemm = [&](const float* A, const _Float16* Wm, const float* b,
                           float* out, int act) {
        int tiles_m = (N + 15) / 16;
        int waves   = tiles_m * 4;          // 4 n-blocks of 64 cols
        int blocks  = (waves + 7) / 8;      // 8 waves (256 threads) per block
        gemm_wmma_kernel<<<blocks, 256, 0, stream>>>(A, Wm, b, out, N, act);
    };

    // 2) shared encoder: h_t = x_t @ enc_W + enc_b
    launch_gemm(x_a, wt, enc_b, h, 0);
    launch_gemm(x_b, wt, enc_b, h + NH, 0);

    float* outp = (float*)d_out;
    for (int l = 0; l < L; ++l) {
        for (int t = 0; t < T; ++t) {
            float*       ht = h + (size_t)t * NH;
            const int*   ei = (t == 0) ? ei_aa : ei_bb;
            const float* at = (t == 0) ? attn_aa : attn_bb;

            // agg = h_t  (residual), then agg[dst] += h_t[src] * attn
            int n4 = (int)(NH / 4);
            copy4_kernel<<<(n4 + 255) / 256, 256, 0, stream>>>(
                (const float4*)ht, (float4*)agg, n4);
            long sthreads = (long)E * 64;
            scatter_kernel<<<(int)((sthreads + 255) / 256), 256, 0, stream>>>(
                (const float4*)ht, ei, ei + E, at, agg, E);

            int mi = l * T + t;
            const _Float16* w1t = wt + (size_t)(1 + mi) * H * H;
            const _Float16* w2t = wt + (size_t)(5 + mi) * H * H;

            // mid = relu(agg @ W1 + b1)
            launch_gemm(agg, w1t, b1 + (size_t)mi * H, mid, 1);

            // h_t (or final output) = mid @ W2 + b2 ; ELU between layers only
            float* dst = (l == L - 1) ? (outp + (size_t)t * NH) : ht;
            launch_gemm(mid, w2t, b2 + (size_t)mi * H, dst, (l < L - 1) ? 2 : 0);
        }
    }
}
